// DiffMAEBottleneckDecoder_2731599200624
// MI455X (gfx1250) — matrix operand order repaired and checked by reading, not compiled, not run on hardware
//
#include <hip/hip_runtime.h>
#include <hip/hip_bf16.h>
#include <stdint.h>

// ---------------------------------------------------------------------------
// MI455X (gfx1250) implementation: all GEMMs via v_wmma_f32_16x16x32_bf16.
// ~740 GFLOP of GEMM vs ~150 MB of traffic @ 23.3 TB/s -> compute bound, so
// everything funnels through the WMMA pipe in bf16 (fp32-range-safe).
// GEMM staging uses CDNA5 async global->LDS copies (ASYNCcnt) with LDS
// double-buffering so tile t+1 copies overlap tile t WMMAs.
// ---------------------------------------------------------------------------

typedef __bf16 bf16_t;
typedef bf16_t v16bf __attribute__((ext_vector_type(16)));
typedef float  v8f   __attribute__((ext_vector_type(8)));

union FragBF { v16bf v; uint32_t u[8]; };

constexpr int BATCH = 64, GROUPS = 16, EIN = 768, EMB = 512, HEADS = 16,
              DHEAD = 32, LAYERS = 8, IDIM = 2048, NP = 196, SEQ = 197,
              KEEP = 49;
constexpr int MTOK   = BATCH * SEQ;        // 12608
constexpr int MEMB   = BATCH * (KEEP + 1); // 3200
constexpr int MPATCH = BATCH * NP;         // 12544

// ---------------------------------------------------------------------------
// Generic bf16 WMMA GEMM: C[M,N] = A[M,K] * Bt[N,K]^T (+bias)(+resid)(+gelu)
// Block tile 128x128, 8 wave32 waves arranged 2(M)x4(N), wave tile 64x32.
// K must be a multiple of 64 (holds for 512/768/2048).
// flags: 1 = exact GELU epilogue, 2 = MAE output (strip CLS row per image).
// ---------------------------------------------------------------------------
__global__ __launch_bounds__(256)
void wmma_gemm(const bf16_t* __restrict__ Ag, const bf16_t* __restrict__ Bt,
               const float* __restrict__ bias, const float* __restrict__ resid,
               float* __restrict__ outF, bf16_t* __restrict__ outH,
               int M, int N, int K, int flags)
{
    constexpr int LDT = 40; // 32 K-elems + 8 pad: 80B rows -> conflict-free frags
    __shared__ bf16_t As[2][128 * LDT];
    __shared__ bf16_t Bs[2][128 * LDT];

    const int tid  = threadIdx.x;
    const int lane = tid & 31, wv = tid >> 5;
    const int wm = wv & 1, wn = wv >> 1;
    const int row = lane & 15, hlf = lane >> 4;
    const int m0 = blockIdx.y * 128, n0 = blockIdx.x * 128;
    const int srow = tid >> 1, sh = tid & 1;

    v8f acc[4][2];
#pragma unroll
    for (int i = 0; i < 4; ++i)
#pragma unroll
        for (int j = 0; j < 2; ++j)
            acc[i][j] = (v8f){0.f, 0.f, 0.f, 0.f, 0.f, 0.f, 0.f, 0.f};

    const bf16_t* gA = Ag + (size_t)(m0 + srow) * K + sh * 16;
    const bf16_t* gB = Bt + (size_t)(n0 + srow) * K + sh * 16;
    const bool arow_ok = (m0 + srow) < M;

    // OOB A rows: pre-zero their LDS slots once; async copies below are
    // EXEC-masked per lane, so these slots stay zero for every K step.
    if (!arow_ok) {
        const uint4 z{0, 0, 0, 0};
#pragma unroll
        for (int bufz = 0; bufz < 2; ++bufz) {
            reinterpret_cast<uint4*>(&As[bufz][srow * LDT + sh * 16])[0]     = z;
            reinterpret_cast<uint4*>(&As[bufz][srow * LDT + sh * 16 + 8])[0] = z;
        }
    }

    // CDNA5 async global->LDS staging: 32B of A + 32B of B per thread per
    // tile, contiguous on both sides so one shared offset:16 covers part 2.
    auto issue_async = [&](int buf, int k0) {
        const uint64_t ga = (uint64_t)(uintptr_t)(gA + k0);
        const uint64_t gb = (uint64_t)(uintptr_t)(gB + k0);
        const uint32_t la =
            (uint32_t)(uintptr_t)(&As[buf][srow * LDT + sh * 16]);
        const uint32_t lb =
            (uint32_t)(uintptr_t)(&Bs[buf][srow * LDT + sh * 16]);
        if (arow_ok) {
            asm volatile("global_load_async_to_lds_b128 %0, %1, off"
                         :: "v"(la), "v"(ga) : "memory");
            asm volatile("global_load_async_to_lds_b128 %0, %1, off offset:16"
                         :: "v"(la), "v"(ga) : "memory");
        }
        asm volatile("global_load_async_to_lds_b128 %0, %1, off"
                     :: "v"(lb), "v"(gb) : "memory");
        asm volatile("global_load_async_to_lds_b128 %0, %1, off offset:16"
                     :: "v"(lb), "v"(gb) : "memory");
    };

    auto compute_tile = [&](int buf) {
        FragBF af[4], bfv[2];
#pragma unroll
        for (int i = 0; i < 4; ++i)
#pragma unroll
            for (int vv = 0; vv < 8; ++vv) {
                // A 16x32 bf16 fragment map: k = (v/4)*16 + half*8 + (v%4)*2
                const int kA = ((vv >> 2) << 4) + (hlf << 3) + ((vv & 3) << 1);
                af[i].u[vv] = *reinterpret_cast<const uint32_t*>(
                    &As[buf][(wm * 64 + i * 16 + row) * LDT + kA]);
            }
#pragma unroll
        for (int j = 0; j < 2; ++j)
#pragma unroll
            for (int vv = 0; vv < 8; ++vv) {
                // B 32x16 bf16 fragment map: k = (v/4)*16 + half*8 + (v%4)*2
                const int kB = ((vv >> 2) << 4) + (hlf << 3) + ((vv & 3) << 1);
                bfv[j].u[vv] = *reinterpret_cast<const uint32_t*>(
                    &Bs[buf][(wn * 32 + j * 16 + row) * LDT + kB]);
            }
#pragma unroll
        for (int i = 0; i < 4; ++i)
#pragma unroll
            for (int j = 0; j < 2; ++j)
                acc[i][j] = __builtin_amdgcn_wmma_f32_16x16x32_bf16(
                    false, af[i].v, false, bfv[j].v, (short)0, acc[i][j],
                    false, false);
    };

#define WAIT_ASYNC() asm volatile("s_wait_asynccnt 0x0" ::: "memory")

    issue_async(0, 0);
    for (int k0 = 0; k0 < K; k0 += 64) {
        WAIT_ASYNC();          // own async copies into buf0 complete
        __syncthreads();       // -> all waves' copies (and zero-fill) visible
        issue_async(1, k0 + 32);
        compute_tile(0);
        WAIT_ASYNC();
        __syncthreads();
        if (k0 + 64 < K) issue_async(0, k0 + 64);
        compute_tile(1);
    }
#undef WAIT_ASYNC

#pragma unroll
    for (int i = 0; i < 4; ++i)
#pragma unroll
        for (int j = 0; j < 2; ++j)
#pragma unroll
            for (int vv = 0; vv < 8; ++vv) {
                const int m = m0 + wm * 64 + i * 16 + vv + hlf * 8;
                const int n = n0 + wn * 32 + j * 16 + row;
                if (m < M) {
                    float c = acc[i][j][vv];
                    if (bias)  c += bias[n];
                    if (resid) c += resid[(size_t)m * N + n];
                    if (flags & 1)
                        c = 0.5f * c * (1.0f + erff(c * 0.70710678118f));
                    if (outF) {
                        if (flags & 2) {
                            const int b = m / SEQ, s = m % SEQ;
                            if (s > 0)
                                outF[((size_t)b * NP + (s - 1)) * N + n] = c;
                        } else {
                            outF[(size_t)m * N + n] = c;
                        }
                    }
                    if (outH) outH[(size_t)m * N + n] = (bf16_t)c;
                }
            }
}

// ---------------------------------------------------------------------------
// Attention: one workgroup per (head, batch). 4 wave32 waves; each wave owns a
// 16-row q-tile. Scores + P*V both via WMMA; register-resident softmax with
// wave32 shfl_xor (reductions stay inside 16-lane halves, matching D layout).
// ---------------------------------------------------------------------------
__global__ __launch_bounds__(128)
void attn_kernel(const bf16_t* __restrict__ Q, const bf16_t* __restrict__ Kk,
                 const bf16_t* __restrict__ Vv, bf16_t* __restrict__ Ctx)
{
    constexpr int KSTR = 40;   // K tile stride (32 dims + pad)
    constexpr int VSTR = 232;  // V^T stride (224 keys + pad)
    constexpr int PSTR = 228;  // probs stride (224 keys + pad)
    __shared__ bf16_t Ks[208 * KSTR];
    __shared__ bf16_t Vt[32 * VSTR];
    __shared__ bf16_t Pb[4][16 * PSTR];

    const int hh = blockIdx.x, b = blockIdx.y;
    const int tid = threadIdx.x;
    const int lane = tid & 31, wv = tid >> 5;
    const int row = lane & 15, hlf = lane >> 4;

    for (int idx = tid; idx < 208 * 32; idx += 128) {
        const int key = idx >> 5, d = idx & 31;
        bf16_t val = (bf16_t)0.f;
        if (key < SEQ)
            val = Kk[((size_t)(b * SEQ + key)) * EMB + hh * DHEAD + d];
        Ks[key * KSTR + d] = val;
    }
    for (int idx = tid; idx < 32 * VSTR; idx += 128) {
        const int d = idx / VSTR, key = idx % VSTR;
        bf16_t val = (bf16_t)0.f;
        if (key < SEQ)
            val = Vv[((size_t)(b * SEQ + key)) * EMB + hh * DHEAD + d];
        Vt[d * VSTR + key] = val;
    }
    __syncthreads();

    const float inv = 0.1767766952966369f; // 1/sqrt(32)
    const v8f vzero = {0.f, 0.f, 0.f, 0.f, 0.f, 0.f, 0.f, 0.f};

    for (int qt0 = 0; qt0 < 16; qt0 += 4) { // uniform trip count for barriers
        const int qt = qt0 + wv;
        const bool active = (qt < 13);
        if (active) {
            FragBF qf;
            const int qrow  = qt * 16 + row;
            const int qrowL = qrow < SEQ ? qrow : SEQ - 1;
#pragma unroll
            for (int vv2 = 0; vv2 < 8; ++vv2) {
                const int kA = ((vv2 >> 2) << 4) + (hlf << 3) + ((vv2 & 3) << 1);
                qf.u[vv2] = *reinterpret_cast<const uint32_t*>(
                    &Q[((size_t)(b * SEQ + qrowL)) * EMB + hh * DHEAD + kA]);
            }
            v8f sc[13];
#pragma unroll
            for (int kt = 0; kt < 13; ++kt) {
                FragBF kf;
#pragma unroll
                for (int vv2 = 0; vv2 < 8; ++vv2) {
                    const int kB = ((vv2 >> 2) << 4) + (hlf << 3) + ((vv2 & 3) << 1);
                    kf.u[vv2] = *reinterpret_cast<const uint32_t*>(
                        &Ks[(kt * 16 + row) * KSTR + kB]);
                }
                sc[kt] = __builtin_amdgcn_wmma_f32_16x16x32_bf16(
                    false, qf.v, false, kf.v, (short)0, vzero, false, false);
            }
#pragma unroll
            for (int kt = 0; kt < 13; ++kt) {
                const bool maskcol = (kt == 12) && (192 + row >= SEQ);
#pragma unroll
                for (int vv2 = 0; vv2 < 8; ++vv2)
                    sc[kt][vv2] = maskcol ? -1e30f : sc[kt][vv2] * inv;
            }
            float mx[8], sum[8];
#pragma unroll
            for (int vv2 = 0; vv2 < 8; ++vv2) {
                float m = sc[0][vv2];
#pragma unroll
                for (int kt = 1; kt < 13; ++kt) m = fmaxf(m, sc[kt][vv2]);
#pragma unroll
                for (int dd = 1; dd < 16; dd <<= 1)
                    m = fmaxf(m, __shfl_xor(m, dd, 32));
                mx[vv2] = m;
            }
#pragma unroll
            for (int vv2 = 0; vv2 < 8; ++vv2) {
                float s = 0.f;
#pragma unroll
                for (int kt = 0; kt < 13; ++kt) {
                    float e = __expf(sc[kt][vv2] - mx[vv2]);
                    sc[kt][vv2] = e;
                    s += e;
                }
#pragma unroll
                for (int dd = 1; dd < 16; dd <<= 1)
                    s += __shfl_xor(s, dd, 32);
                sum[vv2] = 1.0f / s;
            }
#pragma unroll
            for (int vv2 = 0; vv2 < 8; ++vv2) {
                const int r = vv2 + 8 * hlf; // D-layout row
#pragma unroll
                for (int kt = 0; kt < 13; ++kt)
                    Pb[wv][r * PSTR + kt * 16 + row] =
                        (bf16_t)(sc[kt][vv2] * sum[vv2]);
                Pb[wv][r * PSTR + 208 + row] = (bf16_t)0.f; // zero pad tile
            }
        }
        __syncthreads();
        if (active) {
            const int qt16 = qt * 16;
#pragma unroll
            for (int j = 0; j < 2; ++j) {
                v8f co = vzero;
#pragma unroll
                for (int ch = 0; ch < 7; ++ch) {
                    FragBF pf, vf;
#pragma unroll
                    for (int vv2 = 0; vv2 < 8; ++vv2) {
                        const int kA =
                            ((vv2 >> 2) << 4) + (hlf << 3) + ((vv2 & 3) << 1);
                        pf.u[vv2] = *reinterpret_cast<const uint32_t*>(
                            &Pb[wv][row * PSTR + ch * 32 + kA]);
                        const int kB = ((vv2 >> 2) << 4) + (hlf << 3) + ((vv2 & 3) << 1);
                        vf.u[vv2] = *reinterpret_cast<const uint32_t*>(
                            &Vt[(j * 16 + row) * VSTR + ch * 32 + kB]);
                    }
                    co = __builtin_amdgcn_wmma_f32_16x16x32_bf16(
                        false, pf.v, false, vf.v, (short)0, co, false, false);
                }
#pragma unroll
                for (int vv2 = 0; vv2 < 8; ++vv2) {
                    const int qrow2 = qt16 + vv2 + 8 * hlf;
                    if (qrow2 < SEQ)
                        Ctx[((size_t)(b * SEQ + qrow2)) * EMB + hh * DHEAD +
                            j * 16 + row] = (bf16_t)co[vv2];
                }
            }
        }
        __syncthreads();
    }
}

// ---------------------------------------------------------------------------
// LayerNorm over E=512, emits bf16 (feeds next WMMA GEMM as A matrix).
// ---------------------------------------------------------------------------
__global__ __launch_bounds__(128)
void ln_kernel(const float* __restrict__ x, const float* __restrict__ sc,
               const float* __restrict__ bb, bf16_t* __restrict__ out)
{
    const int m = blockIdx.x, t = threadIdx.x;
    const float* xr = x + (size_t)m * EMB;
    float v[4], lsum = 0.f, lsq = 0.f;
#pragma unroll
    for (int i = 0; i < 4; ++i) {
        v[i] = xr[t + i * 128];
        lsum += v[i];
        lsq  += v[i] * v[i];
    }
    for (int dd = 1; dd < 32; dd <<= 1) {
        lsum += __shfl_xor(lsum, dd, 32);
        lsq  += __shfl_xor(lsq, dd, 32);
    }
    __shared__ float ss[4], sq[4];
    const int w = t >> 5;
    if ((t & 31) == 0) { ss[w] = lsum; sq[w] = lsq; }
    __syncthreads();
    const float tsum = ss[0] + ss[1] + ss[2] + ss[3];
    const float tsq  = sq[0] + sq[1] + sq[2] + sq[3];
    const float mu   = tsum * (1.f / EMB);
    const float var  = tsq * (1.f / EMB) - mu * mu;
    const float rstd = rsqrtf(var + 1e-12f);
#pragma unroll
    for (int i = 0; i < 4; ++i) {
        const int e = t + i * 128;
        out[(size_t)m * EMB + e] = (bf16_t)((v[i] - mu) * rstd * sc[e] + bb[e]);
    }
}

// ---------------------------------------------------------------------------
// Small data-movement / pointwise kernels
// ---------------------------------------------------------------------------
__global__ void cvt_copy(const float* __restrict__ s, bf16_t* __restrict__ d,
                         int n)
{
    const int i = blockIdx.x * 256 + threadIdx.x;
    if (i < n) d[i] = (bf16_t)s[i];
}

__global__ void cvt_transpose(const float* __restrict__ s, // [K][N]
                              bf16_t* __restrict__ d,      // [N][K]
                              int K, int N)
{
    const int i = blockIdx.x * 256 + threadIdx.x;
    if (i < K * N) {
        const int k = i / N, n = i % N;
        d[(size_t)n * K + k] = (bf16_t)s[i];
    }
}

__global__ void im2col_kernel(const float* __restrict__ pix,
                              bf16_t* __restrict__ im)
{
    const int idx = blockIdx.x * 256 + threadIdx.x;
    if (idx >= MPATCH * EIN) return;
    const int kk = idx % EIN;
    const int mr = idx / EIN;
    const int b = mr / NP, n = mr % NP;
    const int py = n / 14, px = n % 14;
    const int c = kk >> 8, rr = kk & 255;
    const int ph = rr >> 4, pw = rr & 15;
    im[idx] = (bf16_t)pix[(((size_t)b * 3 + c) * 224 + (py * 16 + ph)) * 224 +
                          (px * 16 + pw)];
}

// h[b,0]=x[b,0]; h[b,1+i]= (r<KEEP ? x[b,1+r] : tok[b,i]); all + pos.
// Uses ids_shuffle[ids_restore[i]] == i (inverse permutation identity).
__global__ void assemble_h(const float* __restrict__ x,
                           const float* __restrict__ tok,
                           const float* __restrict__ pos,
                           const int* __restrict__ idsr,
                           float* __restrict__ h)
{
    const int idx = blockIdx.x * 256 + threadIdx.x;
    if (idx >= MTOK * EMB) return;
    const int e = idx & (EMB - 1);
    const int m = idx >> 9;
    const int b = m / SEQ, sidx = m % SEQ;
    float val;
    if (sidx == 0) {
        val = x[((size_t)b * (KEEP + 1)) * EMB + e];
    } else {
        const int i = sidx - 1;
        const int r = idsr[b * NP + i];
        val = (r < KEEP) ? x[((size_t)b * (KEEP + 1) + 1 + r) * EMB + e]
                         : tok[((size_t)b * NP + i) * EMB + e];
    }
    h[idx] = val + pos[sidx * EMB + e];
}

__global__ void compute_gidx(const int* __restrict__ lengths,
                             int* __restrict__ gidx)
{
    const int b = threadIdx.x;
    if (b >= BATCH) return;
    int acc = 0, g = 0;
    for (int i = 0; i < GROUPS; ++i) {
        const int l = lengths[i];
        if (b >= acc && b < acc + l) g = i;
        acc += l;
    }
    gidx[b] = g;
}

__global__ void group_mean(const float* __restrict__ h,
                           const int* __restrict__ gidx,
                           const int* __restrict__ lengths,
                           float* __restrict__ gmean)
{
    const int g = blockIdx.x;
    const int e = blockIdx.y * 128 + threadIdx.x;
    float s = 0.f;
    for (int b = 0; b < BATCH; ++b)
        if (gidx[b] == g) s += h[((size_t)b * SEQ) * EMB + e];
    gmean[g * EMB + e] = s / (float)lengths[g];
}

__global__ void scatter_cls(float* __restrict__ h, const int* __restrict__ gidx,
                            const float* __restrict__ gmean)
{
    const int b = blockIdx.x;
    const int e = blockIdx.y * 128 + threadIdx.x;
    h[((size_t)b * SEQ) * EMB + e] = gmean[gidx[b] * EMB + e];
}

// ---------------------------------------------------------------------------
// Host orchestration
// ---------------------------------------------------------------------------
extern "C" void kernel_launch(void* const* d_in, const int* in_sizes, int n_in,
                              void* d_out, int out_size, void* d_ws,
                              size_t ws_size, hipStream_t stream)
{
    (void)in_sizes; (void)n_in; (void)out_size; (void)ws_size;
    const float* hs      = (const float*)d_in[0];
    const int*   lengths = (const int*)  d_in[1];
    const int*   idsr    = (const int*)  d_in[2];
    const float* pix     = (const float*)d_in[3];
    const float* embed_w = (const float*)d_in[4];
    const float* embed_b = (const float*)d_in[5];
    const float* pos     = (const float*)d_in[6];
    const float* patch_w = (const float*)d_in[7];
    const float* patch_b = (const float*)d_in[8];
    const float* ln1_s   = (const float*)d_in[9];
    const float* ln1_b   = (const float*)d_in[10];
    const float* wq      = (const float*)d_in[11];
    const float* bq      = (const float*)d_in[12];
    const float* wk      = (const float*)d_in[13];
    const float* bk      = (const float*)d_in[14];
    const float* wvp     = (const float*)d_in[15];
    const float* bv      = (const float*)d_in[16];
    const float* wo      = (const float*)d_in[17];
    const float* bo      = (const float*)d_in[18];
    const float* ln2_s   = (const float*)d_in[19];
    const float* ln2_b   = (const float*)d_in[20];
    const float* wi      = (const float*)d_in[21];
    const float* bi      = (const float*)d_in[22];
    const float* wf      = (const float*)d_in[23];
    const float* bf_     = (const float*)d_in[24];
    const float* lnf_s   = (const float*)d_in[25];
    const float* lnf_b   = (const float*)d_in[26];
    const float* pred_w  = (const float*)d_in[27];
    const float* pred_b  = (const float*)d_in[28];

    char* base = (char*)d_ws;
    size_t off = 0;
    auto alloc = [&](size_t bytes) -> void* {
        off = (off + 255) & ~(size_t)255;
        void* r = base + off;
        off += bytes;
        return r;
    };
    bf16_t* hs_bf = (bf16_t*)alloc((size_t)MEMB * EIN * 2);
    bf16_t* im_bf = (bf16_t*)alloc((size_t)MPATCH * EIN * 2);
    bf16_t* ewT   = (bf16_t*)alloc((size_t)EMB * EIN * 2);
    bf16_t* pwT   = (bf16_t*)alloc((size_t)EMB * EIN * 2);
    bf16_t* wqT   = (bf16_t*)alloc((size_t)LAYERS * EMB * EMB * 2);
    bf16_t* wkT   = (bf16_t*)alloc((size_t)LAYERS * EMB * EMB * 2);
    bf16_t* wvT   = (bf16_t*)alloc((size_t)LAYERS * EMB * EMB * 2);
    bf16_t* woT   = (bf16_t*)alloc((size_t)LAYERS * EMB * EMB * 2);
    bf16_t* wiT   = (bf16_t*)alloc((size_t)LAYERS * IDIM * EMB * 2);
    bf16_t* wfT   = (bf16_t*)alloc((size_t)LAYERS * EMB * IDIM * 2);
    bf16_t* prT   = (bf16_t*)alloc((size_t)EIN * EMB * 2);
    float*  xb    = (float*) alloc((size_t)MEMB * EMB * 4);
    float*  tok   = (float*) alloc((size_t)MPATCH * EMB * 4);
    float*  hb    = (float*) alloc((size_t)MTOK * EMB * 4);
    bf16_t* zb    = (bf16_t*)alloc((size_t)MTOK * EMB * 2);
    bf16_t* qb    = (bf16_t*)alloc((size_t)MTOK * EMB * 2);
    bf16_t* kb    = (bf16_t*)alloc((size_t)MTOK * EMB * 2);
    bf16_t* vb2   = (bf16_t*)alloc((size_t)MTOK * EMB * 2);
    bf16_t* ctxb  = (bf16_t*)alloc((size_t)MTOK * EMB * 2);
    bf16_t* y1    = (bf16_t*)alloc((size_t)MTOK * IDIM * 2);
    int*    gidx  = (int*)   alloc(BATCH * 4);
    float*  gmean = (float*) alloc((size_t)GROUPS * EMB * 4);

    auto cvt = [&](const float* s, bf16_t* d, int n) {
        cvt_copy<<<(n + 255) / 256, 256, 0, stream>>>(s, d, n);
    };
    auto tr = [&](const float* s, bf16_t* d, int Kd, int Nd) {
        cvt_transpose<<<((Kd * Nd) + 255) / 256, 256, 0, stream>>>(s, d, Kd, Nd);
    };
    auto gemm = [&](const bf16_t* A, const bf16_t* Bt, const float* bias,
                    const float* resid, float* oF, bf16_t* oH, int M, int N,
                    int Kd, int flags) {
        dim3 grid(N / 128, (M + 127) / 128);
        wmma_gemm<<<grid, 256, 0, stream>>>(A, Bt, bias, resid, oF, oH, M, N,
                                            Kd, flags);
    };

    // weight conversion to bf16 [N][K]
    cvt(hs, hs_bf, MEMB * EIN);
    cvt(patch_w, pwT, EMB * EIN); // (512,3,16,16) is already [N][K=c*256+ph*16+pw]
    tr(embed_w, ewT, EIN, EMB);
    for (int l = 0; l < LAYERS; ++l) {
        tr(wq  + (size_t)l * EMB * EMB,  wqT + (size_t)l * EMB * EMB,  EMB, EMB);
        tr(wk  + (size_t)l * EMB * EMB,  wkT + (size_t)l * EMB * EMB,  EMB, EMB);
        tr(wvp + (size_t)l * EMB * EMB,  wvT + (size_t)l * EMB * EMB,  EMB, EMB);
        tr(wo  + (size_t)l * EMB * EMB,  woT + (size_t)l * EMB * EMB,  EMB, EMB);
        tr(wi  + (size_t)l * EMB * IDIM, wiT + (size_t)l * EMB * IDIM, EMB, IDIM);
        tr(wf  + (size_t)l * IDIM * EMB, wfT + (size_t)l * IDIM * EMB, IDIM, EMB);
    }
    tr(pred_w, prT, EMB, EIN);

    im2col_kernel<<<(MPATCH * EIN + 255) / 256, 256, 0, stream>>>(pix, im_bf);

    // token embed + patch embed GEMMs
    gemm(hs_bf, ewT, embed_b, nullptr, xb, nullptr, MEMB, EMB, EIN, 0);
    gemm(im_bf, pwT, patch_b, nullptr, tok, nullptr, MPATCH, EMB, EIN, 0);
    compute_gidx<<<1, 64, 0, stream>>>(lengths, gidx);
    assemble_h<<<(MTOK * EMB + 255) / 256, 256, 0, stream>>>(xb, tok, pos, idsr,
                                                             hb);

    for (int l = 0; l < LAYERS; ++l) {
        ln_kernel<<<MTOK, 128, 0, stream>>>(hb, ln1_s + l * EMB,
                                            ln1_b + l * EMB, zb);
        gemm(zb, wqT + (size_t)l * EMB * EMB, bq + l * EMB, nullptr, nullptr,
             qb, MTOK, EMB, EMB, 0);
        gemm(zb, wkT + (size_t)l * EMB * EMB, bk + l * EMB, nullptr, nullptr,
             kb, MTOK, EMB, EMB, 0);
        gemm(zb, wvT + (size_t)l * EMB * EMB, bv + l * EMB, nullptr, nullptr,
             vb2, MTOK, EMB, EMB, 0);
        attn_kernel<<<dim3(HEADS, BATCH), 128, 0, stream>>>(qb, kb, vb2, ctxb);
        gemm(ctxb, woT + (size_t)l * EMB * EMB, bo + l * EMB, hb, hb, nullptr,
             MTOK, EMB, EMB, 0);
        ln_kernel<<<MTOK, 128, 0, stream>>>(hb, ln2_s + l * EMB,
                                            ln2_b + l * EMB, zb);
        gemm(zb, wiT + (size_t)l * IDIM * EMB, bi + l * IDIM, nullptr, nullptr,
             y1, MTOK, IDIM, EMB, 1); // fused exact GELU -> bf16
        gemm(y1, wfT + (size_t)l * IDIM * EMB, bf_ + l * EMB, hb, hb, nullptr,
             MTOK, EMB, IDIM, 0);
        group_mean<<<dim3(GROUPS, EMB / 128), 128, 0, stream>>>(hb, gidx,
                                                                lengths, gmean);
        scatter_cls<<<dim3(BATCH, EMB / 128), 128, 0, stream>>>(hb, gidx,
                                                                gmean);
    }

    ln_kernel<<<MTOK, 128, 0, stream>>>(hb, lnf_s, lnf_b, zb);
    gemm(zb, prT, pred_b, nullptr, (float*)d_out, nullptr, MTOK, EIN, EMB, 2);
}
